// LstmDecoder_73615739453901
// MI455X (gfx1250) — matrix operand order repaired and checked by reading, not compiled, not run on hardware
//
#include <hip/hip_runtime.h>
#include <hip/hip_bf16.h>
#include <math.h>

// ---- problem constants (from reference: B,H,I,O,T = 256,1024,1,512,512) ----
#define BATCH   256
#define HDIM    1024
#define ODIM    512
#define GATES   4096          // 4*H
#define KSL     32            // K per wmma bf16
#define NKS     (HDIM / KSL)  // 32 k-slices
#define NTILE_G (GATES / 16)  // 256 gate column tiles
#define NTILE_O (ODIM / 16)   // 32 head column tiles
#define WGROWS  16            // batch rows per workgroup (WMMA M)
#define NWAVES  16
#define TPB     (NWAVES * 32) // 512 threads
#define TSTRIDE ((size_t)NKS * 32 * 16)   // bf16 elems per packed n-tile
#define KSTRIDE ((size_t)32 * 16)         // bf16 elems per packed k-slice (512)

typedef __attribute__((ext_vector_type(16))) __bf16 v16bf;
typedef __attribute__((ext_vector_type(8)))  __bf16 v8bf;
typedef __attribute__((ext_vector_type(8)))  float  v8f;

// ---------------------------------------------------------------------------
// Pack a row-major fp32 weight W[n][k] (k-stride HDIM) into the per-lane
// bf16 B-operand layout for v_wmma_f32_16x16x32_bf16:
//   lane l of k-slice ks of n-tile t holds 16 contiguous bf16:
//   element e = W[t*16 + (l&15)][ks*32 + (l>>4)*16 + e]
// ---------------------------------------------------------------------------
__global__ void pack_b_kernel(const float* __restrict__ w, __bf16* __restrict__ p) {
    int tile = blockIdx.x >> 5;       // /NKS
    int ks   = blockIdx.x & 31;       // %NKS
    int lane = threadIdx.x;
    int n    = tile * 16 + (lane & 15);
    int k0   = ks * KSL + ((lane >> 4) << 4);
    const float* src = w + (size_t)n * HDIM + k0;
    __bf16* dst = p + ((((size_t)tile * NKS + ks) * 32) + lane) * 16;
#pragma unroll
    for (int e = 0; e < 16; ++e) dst[e] = (__bf16)src[e];
}

__device__ __forceinline__ float sigm(float x) {
    return 1.0f / (1.0f + __expf(-x));
}

// A-operand loader: 16x32 bf16 tile of h from LDS.
// lane l (row = l&15) holds K = kb..kb+15, kb = (l>=16)?16:0.
__device__ __forceinline__ v16bf load_a(const __bf16* hbuf, int lane, int ks) {
    int row = lane & 15;
    int kb  = ks * KSL + ((lane >> 4) << 4);
    const v8bf* p0 = (const v8bf*)(hbuf + row * HDIM + kb);
    const v8bf* p1 = (const v8bf*)(hbuf + row * HDIM + kb + 8);
    v8bf lo = *p0, hi = *p1;
    v16bf a;
#pragma unroll
    for (int j = 0; j < 8; ++j) { a[j] = lo[j]; a[j + 8] = hi[j]; }
    return a;
}

// ---------------------------------------------------------------------------
// Persistent LSTM decoder: 16 workgroups x 16 batch rows, all T steps.
// Dynamic LDS: bf16 h double buffer [2][16][1024] = 64KB.
// ---------------------------------------------------------------------------
__global__ void __launch_bounds__(TPB)
lstm_decoder_kernel(const float* __restrict__ z,
                    const float* __restrict__ w_ih,
                    const float* __restrict__ b_ih,
                    const float* __restrict__ b_hh,
                    const float* __restrict__ b_lin,
                    const __bf16* __restrict__ pWhh,
                    const __bf16* __restrict__ pWlin,
                    const int*   __restrict__ seqlen,
                    float*       __restrict__ out) {
    extern __shared__ char smem_raw[];
    __bf16* h_lds = (__bf16*)smem_raw;             // [2][WGROWS][HDIM]
    __shared__ float head_lds[WGROWS][ODIM];       // 32KB logits staging
    __shared__ float x_lds[WGROWS];                // greedy-feedback scalar/row

    const int tid  = threadIdx.x;
    const int lane = tid & 31;
    const int wv   = tid >> 5;                     // wave id 0..15
    const int b0   = blockIdx.x * WGROWS;
    const int T    = seqlen[0];
    const int halfsel = (lane >> 4);               // 0 for lanes 0-15, 1 for 16-31
    const int ncol = lane & 15;

    // init: h0 = z (bf16 into buffer 0), x0 = 0, c0 = 0 (registers)
    for (int i = tid; i < WGROWS * HDIM; i += TPB) {
        int r = i >> 10, k = i & (HDIM - 1);
        h_lds[r * HDIM + k] = (__bf16)z[(size_t)(b0 + r) * HDIM + k];
    }
    if (tid < WGROWS) x_lds[tid] = 0.0f;

    v8f c_t[4] = {};                               // cell state, 4 col-tiles/wave
    __syncthreads();

    for (int t = 0; t < T; ++t) {
        const __bf16* hc = h_lds + (size_t)(t & 1) * (WGROWS * HDIM);
        __bf16*       hn = h_lds + (size_t)((t & 1) ^ 1) * (WGROWS * HDIM);

        // Launder a zero *offset* (not the pointer): keeps the weight base as a
        // promoted kernarg pointer (=> global_load_b128, LOADcnt only, no FLAT
        // DScnt coupling) while making the packed-B loads non-invariant across
        // the t loop so LICM can't hoist ~16KB/lane of weights into spills.
        size_t zoff = 0;
        asm volatile("" : "+s"(zoff));
        const __bf16* pwh = pWhh + zoff;
        const __bf16* pwl = pWlin + zoff;

        // ---- gates GEMM + cell update, chunk m handles h-col tile p=wv+16m ----
#pragma unroll
        for (int m = 0; m < 4; ++m) {
            const int p = wv + 16 * m;             // h column tile 0..63
            const __bf16* base0 = pwh + (size_t)(0 * 64 + p) * TSTRIDE + (size_t)lane * 16;
            const __bf16* base1 = pwh + (size_t)(1 * 64 + p) * TSTRIDE + (size_t)lane * 16;
            const __bf16* base2 = pwh + (size_t)(2 * 64 + p) * TSTRIDE + (size_t)lane * 16;
            const __bf16* base3 = pwh + (size_t)(3 * 64 + p) * TSTRIDE + (size_t)lane * 16;
            v8f acc0 = {}, acc1 = {}, acc2 = {}, acc3 = {};
            // software-pipelined B: prefetch k-slice ks+1 while WMMAs run on ks
            v16bf b0 = *(const v16bf*)(base0);
            v16bf b1 = *(const v16bf*)(base1);
            v16bf b2 = *(const v16bf*)(base2);
            v16bf b3 = *(const v16bf*)(base3);
#pragma unroll 4
            for (int ks = 0; ks < NKS; ++ks) {
                const size_t nko = (size_t)((ks + 1 < NKS) ? ks + 1 : ks) * KSTRIDE;
                v16bf a  = load_a(hc, lane, ks);
                v16bf n0 = *(const v16bf*)(base0 + nko);
                v16bf n1 = *(const v16bf*)(base1 + nko);
                v16bf n2 = *(const v16bf*)(base2 + nko);
                v16bf n3 = *(const v16bf*)(base3 + nko);
                acc0 = __builtin_amdgcn_wmma_f32_16x16x32_bf16(false, a, false, b0, (short)0, acc0, false, false);
                acc1 = __builtin_amdgcn_wmma_f32_16x16x32_bf16(false, a, false, b1, (short)0, acc1, false, false);
                acc2 = __builtin_amdgcn_wmma_f32_16x16x32_bf16(false, a, false, b2, (short)0, acc2, false, false);
                acc3 = __builtin_amdgcn_wmma_f32_16x16x32_bf16(false, a, false, b3, (short)0, acc3, false, false);
                b0 = n0; b1 = n1; b2 = n2; b3 = n3;
            }
            // elementwise LSTM cell update for h columns in tile p
            const int col = p * 16 + ncol;         // h column index, fixed per lane
            float bia = b_ih[0 * HDIM + col] + b_hh[0 * HDIM + col];
            float bfa = b_ih[1 * HDIM + col] + b_hh[1 * HDIM + col];
            float bga = b_ih[2 * HDIM + col] + b_hh[2 * HDIM + col];
            float boa = b_ih[3 * HDIM + col] + b_hh[3 * HDIM + col];
            float wia = w_ih[0 * HDIM + col];
            float wfa = w_ih[1 * HDIM + col];
            float wga = w_ih[2 * HDIM + col];
            float woa = w_ih[3 * HDIM + col];
#pragma unroll
            for (int r = 0; r < 8; ++r) {
                int row = r + (halfsel << 3);
                float xb = x_lds[row];
                float gi = acc0[r] + bia + xb * wia;
                float gf = acc1[r] + bfa + xb * wfa;
                float gg = acc2[r] + bga + xb * wga;
                float go = acc3[r] + boa + xb * woa;
                float cv = sigm(gf) * c_t[m][r] + sigm(gi) * tanhf(gg);
                c_t[m][r] = cv;
                float hv = sigm(go) * tanhf(cv);
                hn[row * HDIM + col] = (__bf16)hv;  // new h for next step / head
            }
        }
        __syncthreads();   // new h complete before head GEMM reads it

        // ---- head GEMM: x_t = h_new @ w_lin^T + b_lin, 2 tiles per wave ----
#pragma unroll
        for (int s = 0; s < 2; ++s) {
            const int tile = 2 * wv + s;           // 0..31
            const __bf16* baseh = pwl + (size_t)tile * TSTRIDE + (size_t)lane * 16;
            v8f acc = {};
            v16bf b = *(const v16bf*)(baseh);
#pragma unroll 4
            for (int ks = 0; ks < NKS; ++ks) {
                const size_t nko = (size_t)((ks + 1 < NKS) ? ks + 1 : ks) * KSTRIDE;
                v16bf a = load_a(hn, lane, ks);
                v16bf n = *(const v16bf*)(baseh + nko);
                acc = __builtin_amdgcn_wmma_f32_16x16x32_bf16(false, a, false, b, (short)0, acc, false, false);
                b = n;
            }
            const int col = tile * 16 + ncol;
            float bl = b_lin[col];
#pragma unroll
            for (int r = 0; r < 8; ++r) {
                int row = r + (halfsel << 3);
                head_lds[row][col] = acc[r] + bl;
            }
        }
        __syncthreads();   // logits staged

        // ---- greedy feedback: wave wv computes argmax of row wv (first max) ----
        {
            float bv = -INFINITY; int bidx = 0;
#pragma unroll
            for (int i = 0; i < ODIM / 32; ++i) {
                int cidx = lane + i * 32;
                float v = head_lds[wv][cidx];
                if (v > bv) { bv = v; bidx = cidx; }   // ascending scan keeps first
            }
            for (int off = 16; off > 0; off >>= 1) {
                float ov = __shfl_down(bv, off, 32);
                int   oi = __shfl_down(bidx, off, 32);
                if (ov > bv || (ov == bv && oi < bidx)) { bv = ov; bidx = oi; }
            }
            if (lane == 0) x_lds[wv] = (float)bidx;
        }

        // ---- coalesced store of logits [16][512] to out[b, t, :] ----
#pragma unroll
        for (int i = 0; i < (WGROWS * ODIM) / TPB; ++i) {
            int e = tid + i * TPB;
            int row = e >> 9, col = e & (ODIM - 1);
            out[(((size_t)(b0 + row)) * T + t) * ODIM + col] = head_lds[row][col];
        }
        __syncthreads();   // x_lds ready; head_lds reusable next step
    }
}

// ---------------------------------------------------------------------------
extern "C" void kernel_launch(void* const* d_in, const int* in_sizes, int n_in,
                              void* d_out, int out_size, void* d_ws, size_t ws_size,
                              hipStream_t stream) {
    const float* z     = (const float*)d_in[0];   // [B,H]
    const float* w_ih  = (const float*)d_in[1];   // [4H,1]
    const float* w_hh  = (const float*)d_in[2];   // [4H,H]
    const float* b_ih  = (const float*)d_in[3];   // [4H]
    const float* b_hh  = (const float*)d_in[4];   // [4H]
    const float* w_lin = (const float*)d_in[5];   // [O,H]
    const float* b_lin = (const float*)d_in[6];   // [O]
    const int*   seqT  = (const int*)d_in[7];     // scalar T
    float* out = (float*)d_out;

    // workspace: packed bf16 B-operands (w_hh 8MB, w_lin 1MB)
    __bf16* pWhh  = (__bf16*)d_ws;
    __bf16* pWlin = (__bf16*)((char*)d_ws + (size_t)GATES * HDIM * sizeof(__bf16));

    pack_b_kernel<<<NTILE_G * NKS, 32, 0, stream>>>(w_hh, pWhh);
    pack_b_kernel<<<NTILE_O * NKS, 32, 0, stream>>>(w_lin, pWlin);

    const size_t dyn_lds = (size_t)2 * WGROWS * HDIM * sizeof(__bf16); // 64KB h dbl-buf
    lstm_decoder_kernel<<<BATCH / WGROWS, TPB, dyn_lds, stream>>>(
        z, w_ih, b_ih, b_hh, b_lin, pWhh, pWlin, seqT, out);
}
